// CausalSelfAttention_11373073399841
// MI455X (gfx1250) — compile-verified, matrix operand order checked by reading, not run on hardware
//
#include <hip/hip_runtime.h>
#include <hip/hip_bf16.h>
#include <math.h>

typedef __bf16 bf16;
typedef __attribute__((ext_vector_type(8)))  __bf16 v8bf;
typedef __attribute__((ext_vector_type(16))) __bf16 v16bf;
typedef __attribute__((ext_vector_type(8)))  float  v8f;
typedef __attribute__((ext_vector_type(4)))  uint32_t u32x4;
typedef __attribute__((ext_vector_type(8)))  uint32_t u32x8;

#define B_  4
#define T_  2048
#define C_  2048
#define NH  16
#define NKV 4
#define HD  128

// Build a 16x32 bf16 A/B fragment per CDNA5 ISA layout:
// lane L: row/col = L&15, K elements = [koff..koff+7] and [koff+16..koff+23]
// where koff = (L>>4)*8, from a contiguous row at p (caller adds row*ld + koff).
static __device__ __forceinline__ v16bf frag16(const bf16* p) {
    v8bf lo = *(const v8bf*)(p);
    v8bf hi = *(const v8bf*)(p + 16);
    return __builtin_shufflevector(lo, hi, 0,1,2,3,4,5,6,7,8,9,10,11,12,13,14,15);
}

static __device__ __forceinline__ v8f wmma_bf16(v16bf a, v16bf b, v8f c) {
    return __builtin_amdgcn_wmma_f32_16x16x32_bf16(false, a, false, b, (short)0, c, false, false);
}

// LDS byte offset of a generic shared pointer: HW maps flat->LDS via addr[31:0].
static __device__ __forceinline__ uint32_t lds_off(const void* p) {
    return (uint32_t)(uintptr_t)p;
}

// gfx1250 async DMA: 16B/lane global -> LDS, tracked by ASYNCcnt.
static __device__ __forceinline__ void async_g2l_b128(uint32_t lds_byte_addr,
                                                      const bf16* gaddr) {
    asm volatile("global_load_async_to_lds_b128 %0, %1, off"
                 :: "v"(lds_byte_addr), "v"(gaddr) : "memory");
}

// Tensor Data Mover: one wave-issued DMA of a 2D tile (rows x cols, bf16) from
// global (row stride = stride elems) into LDS, packed dense. Tracked by TENSORcnt.
// D# fields per CDNA5 ISA 8.3/8.4 (count=1, type=2, data_size=2B, 2D tile).
static __device__ __forceinline__ void tdm_tile_2d(uint32_t lds_byte_addr,
                                                   const bf16* gtile,
                                                   uint32_t cols, uint32_t rows,
                                                   uint32_t stride,
                                                   uint32_t tdim0, uint32_t tdim1) {
    uint64_t ga = (uint64_t)(uintptr_t)gtile;
    u32x4 g0;
    g0[0] = 1u;                                        // count=1, user mode
    g0[1] = lds_byte_addr;                             // lds_addr [63:32]
    g0[2] = (uint32_t)ga;                              // global_addr [95:64]
    g0[3] = (uint32_t)(ga >> 32) | (2u << 30);         // global_addr hi | type=2
    u32x8 g1;
    g1[0] = 1u << 16;                                  // wg_mask=0, data_size=2B
    g1[1] = (tdim0 & 0xffffu) << 16;                   // tensor_dim0 lo16 [63:48]
    g1[2] = (tdim0 >> 16) | ((tdim1 & 0xffffu) << 16); // tensor_dim0 hi | dim1 lo
    g1[3] = (tdim1 >> 16) | (cols << 16);              // tensor_dim1 hi | tile_dim0
    g1[4] = rows;                                      // tile_dim1 (tile_dim2=0)
    g1[5] = stride;                                    // tensor_dim0_stride lo32
    g1[6] = 0u;                                        // stride hi | dim1_stride lo
    g1[7] = 0u;
    asm volatile("tensor_load_to_lds %0, %1" :: "s"(g0), "s"(g1) : "memory");
}

__global__ __launch_bounds__(256) void cvt_f32_to_bf16(const float* __restrict__ in,
                                                       bf16* __restrict__ out, int n) {
    int i = blockIdx.x * 256 + threadIdx.x;
    if (i < n) out[i] = (bf16)in[i];
}

// W: [K,N] f32 row-major -> Wt: [N,K] bf16 (so GEMM B-tiles are contiguous rows).
__global__ __launch_bounds__(256) void cvt_transpose_w(const float* __restrict__ W,
                                                       bf16* __restrict__ Wt,
                                                       int K, int N) {
    size_t i = (size_t)blockIdx.x * 256 + threadIdx.x;
    if (i >= (size_t)K * N) return;
    int n = (int)(i % (size_t)N);
    int k = (int)(i / (size_t)N);
    Wt[(size_t)n * K + k] = (bf16)W[i];
}

// vb: [B*T, NKV*HD] row-major  ->  vT: [(b*NKV+h)*HD + d][T]
__global__ __launch_bounds__(256) void transpose_v_k(const bf16* __restrict__ v,
                                                     bf16* __restrict__ vT) {
    size_t i = (size_t)blockIdx.x * 256 + threadIdx.x;
    if (i >= (size_t)B_ * T_ * NKV * HD) return;
    int d = (int)(i & (HD - 1));
    int h = (int)((i >> 7) & (NKV - 1));
    size_t bt = i >> 9;
    int t = (int)(bt & (T_ - 1));
    int b = (int)(bt >> 11);
    vT[(((size_t)(b * NKV + h) * HD + d) << 11) + t] = v[i];
}

// C[M,N] = A[M,K] @ Bt[N,K]^T, bf16 in, f32 accumulate, bf16 or f32 out.
// Block: 256 thr = 8 waves; tile 128x128x32; wave tile 64x32 (4x2 WMMA frags).
// A-tile staged by the Tensor Data Mover (wave 0, TENSORcnt); B-tile staged by
// per-lane async global->LDS b128 copies (ASYNCcnt). Double-buffered.
__global__ __launch_bounds__(256) void gemm_bf16_k(
    const bf16* __restrict__ A, const bf16* __restrict__ Bt,
    bf16* __restrict__ Cb, float* __restrict__ Cf,
    int M, int N, int K) {
    __shared__ __align__(16) bf16 ldsA[2][128 * 32];   // [m][k]
    __shared__ __align__(16) bf16 ldsB[2][128 * 32];   // [n][k]
    const int tid  = threadIdx.x;
    const int lane = tid & 31;
    const int w    = tid >> 5;
    const int wr   = w >> 2, wc = w & 3;
    const int m0   = blockIdx.y * 128;
    const int n0   = blockIdx.x * 128;
    const int row  = lane & 15;
    const int koff = (lane >> 4) << 3;  // 0 or 8

    // This thread's two 16B chunks within the 128x32 B tile.
    const int c0 = tid, c1 = tid + 256;
    const int rb0 = c0 >> 2, kb0 = (c0 & 3) << 3;
    const int rb1 = c1 >> 2, kb1 = (c1 & 3) << 3;

    auto stageA = [&](int buf, int kk) {  // TDM, call from wave 0 only
        tdm_tile_2d(lds_off(&ldsA[buf][0]),
                    A + (size_t)m0 * K + kk,
                    /*cols=*/32, /*rows=*/128, /*stride=*/(uint32_t)K,
                    /*tdim0=*/(uint32_t)K, /*tdim1=*/128);
    };
    auto stageB = [&](int buf, int kk) {  // per-lane async copies
        async_g2l_b128(lds_off(&ldsB[buf][rb0 * 32 + kb0]),
                       &Bt[(size_t)(n0 + rb0) * K + kk + kb0]);
        async_g2l_b128(lds_off(&ldsB[buf][rb1 * 32 + kb1]),
                       &Bt[(size_t)(n0 + rb1) * K + kk + kb1]);
    };

    v8f acc[4][2] = {};
    const int nk = K >> 5;
    if (w == 0) stageA(0, 0);
    stageB(0, 0);
    for (int t = 0; t < nk; ++t) {
        const int cur = t & 1;
        if (t + 1 < nk) {
            if (w == 0) stageA(cur ^ 1, (t + 1) << 5);
            stageB(cur ^ 1, (t + 1) << 5);
            if (w == 0) __builtin_amdgcn_s_wait_tensorcnt(1);  // cur A landed
            asm volatile("s_wait_asynccnt 0x2" ::: "memory");  // cur B landed
        } else {
            if (w == 0) __builtin_amdgcn_s_wait_tensorcnt(0);
            asm volatile("s_wait_asynccnt 0x0" ::: "memory");
        }
        __syncthreads();

        v16bf af[4], bfrg[2];
        #pragma unroll
        for (int mi = 0; mi < 4; ++mi)
            af[mi] = frag16(&ldsA[cur][(wr * 64 + mi * 16 + row) * 32 + koff]);
        #pragma unroll
        for (int nj = 0; nj < 2; ++nj)
            bfrg[nj] = frag16(&ldsB[cur][(wc * 32 + nj * 16 + row) * 32 + koff]);
        #pragma unroll
        for (int mi = 0; mi < 4; ++mi)
            #pragma unroll
            for (int nj = 0; nj < 2; ++nj)
                acc[mi][nj] = wmma_bf16(af[mi], bfrg[nj], acc[mi][nj]);
        __syncthreads();  // all waves done reading cur before it is re-staged
    }
    // Epilogue: C frag layout — VGPR e: row e (lanes 0-15) / e+8 (lanes 16-31).
    #pragma unroll
    for (int mi = 0; mi < 4; ++mi)
        #pragma unroll
        for (int nj = 0; nj < 2; ++nj)
            #pragma unroll
            for (int e = 0; e < 8; ++e) {
                int r = m0 + wr * 64 + mi * 16 + e + koff;
                int cidx = n0 + wc * 32 + nj * 16 + row;
                float val = acc[mi][nj][e];
                if (Cf) Cf[(size_t)r * N + cidx] = val;
                else    Cb[(size_t)r * N + cidx] = (bf16)val;
            }
}

// Flash attention: block = 8 waves, 128 query rows; wave owns 16 rows.
// Streams 32 keys/iter: S via 8 WMMAs, online softmax, O += P@V via 8 WMMAs.
__global__ __launch_bounds__(256) void attn_k(
    const bf16* __restrict__ Q, const bf16* __restrict__ Kb,
    const bf16* __restrict__ VT, bf16* __restrict__ Y) {
    __shared__ __align__(16) bf16 ldsP[8][16 * 32];  // per-wave P tile [16 q][32 k]
    const int tid  = threadIdx.x;
    const int lane = tid & 31;
    const int w    = tid >> 5;
    const int bh   = blockIdx.y;
    const int b    = bh >> 4;
    const int h    = bh & 15;
    const int hkv  = h >> 2;
    const int q0   = blockIdx.x * 128 + w * 16;
    const int row  = lane & 15;
    const int koff = (lane >> 4) << 3;
    const float scl   = 0.08838834764831845f;                  // 1/sqrt(128)
    const float slope = exp2f(-0.5f * (float)((h >> 2) + 1));  // ALiBi, GQA-repeated

    // Q fragments (16 rows x 128 dims = 4 K-chunks), straight from global.
    v16bf qf[4];
    const bf16* qp = Q + (size_t)(b * T_ + q0 + row) * C_ + h * HD + koff;
    #pragma unroll
    for (int c = 0; c < 4; ++c) qf[c] = frag16(qp + c * 32);

    v8f O[8] = {};
    float mrun[8], lrun[8];
    #pragma unroll
    for (int e = 0; e < 8; ++e) { mrun[e] = -1e30f; lrun[e] = 0.f; }

    const int kmax = q0 + 15;
    for (int jt = 0; jt <= kmax; jt += 32) {
        // S = Q @ K^T for 32 keys (two 16-key fragments).
        v8f S0 = {}, S1 = {};
        const bf16* kp0 = Kb + (size_t)(b * T_ + jt + row) * (NKV * HD) + hkv * HD + koff;
        const bf16* kp1 = kp0 + (size_t)16 * (NKV * HD);
        #pragma unroll
        for (int c = 0; c < 4; ++c) {
            S0 = wmma_bf16(qf[c], frag16(kp0 + c * 32), S0);
            S1 = wmma_bf16(qf[c], frag16(kp1 + c * 32), S1);
        }
        // Online softmax per row (row e+koff held by one 16-lane half).
        #pragma unroll
        for (int e = 0; e < 8; ++e) {
            int qi = q0 + e + koff;
            int j0 = jt + row;
            int j1 = jt + 16 + row;
            float s0 = (j0 <= qi) ? (S0[e] * scl + slope * (float)(j0 - qi)) : -1e30f;
            float s1 = (j1 <= qi) ? (S1[e] * scl + slope * (float)(j1 - qi)) : -1e30f;
            float vmx = fmaxf(s0, s1);
            #pragma unroll
            for (int off = 1; off < 16; off <<= 1) vmx = fmaxf(vmx, __shfl_xor(vmx, off, 32));
            float mn   = fmaxf(mrun[e], vmx);
            float corr = __expf(mrun[e] - mn);
            float p0   = __expf(s0 - mn);
            float p1   = __expf(s1 - mn);
            float rs   = p0 + p1;
            #pragma unroll
            for (int off = 1; off < 16; off <<= 1) rs += __shfl_xor(rs, off, 32);
            lrun[e] = lrun[e] * corr + rs;
            mrun[e] = mn;
            #pragma unroll
            for (int f = 0; f < 8; ++f) O[f][e] *= corr;
            ldsP[w][(e + koff) * 32 + row]      = (bf16)p0;
            ldsP[w][(e + koff) * 32 + row + 16] = (bf16)p1;
        }
        __builtin_amdgcn_wave_barrier();  // order intra-wave LDS store->load
        // O += P @ V : P as 16x32 A-frag from LDS, V as 32x16 B-frags from vT.
        v16bf pf = frag16(&ldsP[w][row * 32 + koff]);
        #pragma unroll
        for (int f = 0; f < 8; ++f) {
            const bf16* vp = VT + ((size_t)((b * NKV + hkv) * HD + f * 16 + row) << 11)
                             + jt + koff;
            O[f] = wmma_bf16(pf, frag16(vp), O[f]);
        }
    }
    // Normalize and write y[b,t, h*128 + d] as bf16 for the out-projection.
    #pragma unroll
    for (int f = 0; f < 8; ++f)
        #pragma unroll
        for (int e = 0; e < 8; ++e) {
            int qi = q0 + e + koff;
            float inv = 1.0f / lrun[e];
            Y[(size_t)(b * T_ + qi) * C_ + h * HD + f * 16 + row] = (bf16)(O[f][e] * inv);
        }
}

extern "C" void kernel_launch(void* const* d_in, const int* in_sizes, int n_in,
                              void* d_out, int out_size, void* d_ws, size_t ws_size,
                              hipStream_t stream) {
    (void)in_sizes; (void)n_in; (void)out_size; (void)ws_size;
    const float* x  = (const float*)d_in[0];
    const float* Wq = (const float*)d_in[1];
    const float* Wk = (const float*)d_in[2];
    const float* Wv = (const float*)d_in[3];
    const float* Wo = (const float*)d_in[4];
    float* out = (float*)d_out;

    char* ws = (char*)d_ws;
    size_t off = 0;
    auto alloc = [&](size_t bytes) {
        char* p = ws + off;
        off += (bytes + 255) & ~(size_t)255;
        return p;
    };
    const size_t MT = (size_t)B_ * T_;  // 8192
    bf16* xb  = (bf16*)alloc(MT * C_ * 2);
    bf16* wqt = (bf16*)alloc((size_t)C_ * C_ * 2);        // [N=2048][K=2048]
    bf16* wkt = (bf16*)alloc((size_t)C_ * NKV * HD * 2);  // [N=512][K=2048]
    bf16* wvt = (bf16*)alloc((size_t)C_ * NKV * HD * 2);
    bf16* wot = (bf16*)alloc((size_t)C_ * C_ * 2);
    bf16* qb  = (bf16*)alloc(MT * C_ * 2);
    bf16* kb  = (bf16*)alloc(MT * NKV * HD * 2);
    bf16* vb  = (bf16*)alloc(MT * NKV * HD * 2);
    bf16* vT  = (bf16*)alloc(MT * NKV * HD * 2);
    bf16* yb  = (bf16*)alloc(MT * C_ * 2);

    // x -> bf16 (row-major, GEMM A side)
    hipLaunchKernelGGL(cvt_f32_to_bf16, dim3((unsigned)((MT * C_ + 255) / 256)), dim3(256),
                       0, stream, x, xb, (int)(MT * C_));
    // Weights -> bf16, transposed to [N][K] for contiguous async B-tiles.
    auto cvtW = [&](const float* Wm, bf16* Wt, int K, int N) {
        hipLaunchKernelGGL(cvt_transpose_w, dim3((unsigned)(((size_t)K * N + 255) / 256)),
                           dim3(256), 0, stream, Wm, Wt, K, N);
    };
    cvtW(Wq, wqt, C_, C_);
    cvtW(Wk, wkt, C_, NKV * HD);
    cvtW(Wv, wvt, C_, NKV * HD);
    cvtW(Wo, wot, C_, C_);

    // Projections: q = x@Wq (8192x2048x2048), k/v = x@Wk/Wv (8192x512x2048)
    hipLaunchKernelGGL(gemm_bf16_k, dim3(C_ / 128, (unsigned)(MT / 128)), dim3(256), 0, stream,
                       xb, wqt, qb, (float*)nullptr, (int)MT, C_, C_);
    hipLaunchKernelGGL(gemm_bf16_k, dim3((NKV * HD) / 128, (unsigned)(MT / 128)), dim3(256), 0, stream,
                       xb, wkt, kb, (float*)nullptr, (int)MT, NKV * HD, C_);
    hipLaunchKernelGGL(gemm_bf16_k, dim3((NKV * HD) / 128, (unsigned)(MT / 128)), dim3(256), 0, stream,
                       xb, wvt, vb, (float*)nullptr, (int)MT, NKV * HD, C_);

    hipLaunchKernelGGL(transpose_v_k, dim3((unsigned)((MT * NKV * HD + 255) / 256)), dim3(256),
                       0, stream, vb, vT);

    hipLaunchKernelGGL(attn_k, dim3(T_ / 128, B_ * NH), dim3(256), 0, stream, qb, kb, vT, yb);

    // Out projection: out = y @ Wo -> f32 d_out
    hipLaunchKernelGGL(gemm_bf16_k, dim3(C_ / 128, (unsigned)(MT / 128)), dim3(256), 0, stream,
                       yb, wot, (bf16*)nullptr, out, (int)MT, C_, C_);
}
